// MolecularModel_8839042695342
// MI455X (gfx1250) — matrix operand order repaired and checked by reading, not compiled, not run on hardware
//
#include <hip/hip_runtime.h>
#include <hip/hip_bf16.h>
#include <math.h>

// ---------------------------------------------------------------------------
// Model dims (compile-time constants matching the reference)
// ---------------------------------------------------------------------------
#define NN      8192        // nodes / sequence length
#define EE      262144      // edges
#define D_NODE  128
#define D_EDGE  10
#define D_MSG   64
#define H_LSTM  256
#define D_ATT   512         // 2*H_LSTM
#define G4H     1024        // 4*H_LSTM

typedef __attribute__((ext_vector_type(16))) _Float16 v16h;
typedef __attribute__((ext_vector_type(8)))  _Float16 v8h;
typedef __attribute__((ext_vector_type(8)))  float    v8f;

// ---------------------------------------------------------------------------
// 1) MPNN message + scatter-add (fused).  One thread per (edge, out-channel).
// ---------------------------------------------------------------------------
__global__ void mpnn_msg_scatter(const float* __restrict__ x,
                                 const int*   __restrict__ ei,   // [2,E]
                                 const float* __restrict__ ea,   // [E,10]
                                 const float* __restrict__ mw,   // [64,138]
                                 const float* __restrict__ mb,   // [64]
                                 float* __restrict__ aggr)       // [N,64]
{
    long gid = (long)blockIdx.x * blockDim.x + threadIdx.x;
    int e = (int)(gid >> 6);
    int o = (int)(gid & 63);
    if (e >= EE) return;
    int src = ei[e];
    int dst = ei[EE + e];
    const float* xr = x + (size_t)src * D_NODE;
    const float* wr = mw + (size_t)o * (D_NODE + D_EDGE);
    float acc = mb[o];
#pragma unroll 8
    for (int k = 0; k < D_NODE; ++k) acc += wr[k] * xr[k];
    const float* er = ea + (size_t)e * D_EDGE;
#pragma unroll
    for (int k = 0; k < D_EDGE; ++k) acc += wr[D_NODE + k] * er[k];
    atomicAdd(&aggr[(size_t)dst * D_MSG + o], acc);
}

// ---------------------------------------------------------------------------
// 2) MPNN update: h = [x, aggr] @ upd_w.T + upd_b  -> written as f16
// ---------------------------------------------------------------------------
__global__ void mpnn_update(const float* __restrict__ x,
                            const float* __restrict__ aggr,
                            const float* __restrict__ uw,   // [128,192]
                            const float* __restrict__ ub,   // [128]
                            _Float16* __restrict__ h16)     // [N,128]
{
    long gid = (long)blockIdx.x * blockDim.x + threadIdx.x;
    int n = (int)(gid >> 7);
    int o = (int)(gid & 127);
    if (n >= NN) return;
    const float* wr = uw + (size_t)o * (D_NODE + D_MSG);
    const float* xr = x + (size_t)n * D_NODE;
    const float* ar = aggr + (size_t)n * D_MSG;
    float acc = ub[o];
#pragma unroll 8
    for (int k = 0; k < D_NODE; ++k) acc += wr[k] * xr[k];
#pragma unroll 8
    for (int k = 0; k < D_MSG; ++k) acc += wr[D_NODE + k] * ar[k];
    h16[(size_t)n * D_NODE + o] = (_Float16)acc;
}

// ---------------------------------------------------------------------------
// f32 -> f16 conversion
// ---------------------------------------------------------------------------
__global__ void cvt_f32_f16(const float* __restrict__ in, _Float16* __restrict__ out, long n)
{
    long i = (long)blockIdx.x * blockDim.x + threadIdx.x;
    if (i < n) out[i] = (_Float16)in[i];
}

// ---------------------------------------------------------------------------
// 3) Register-blocked WMMA GEMM:
//      C[M,N] = scale * (A[M,K] @ Bt[N,K]^T) + bias[N]
//    One wave computes a 32x64 block = 2x4 tiles of 16x16
//    -> 8 v_wmma_f32_16x16x32_f16 per K-step fed by 6 fragment loads
//       (2.7x less L2 traffic than 1 tile/wave; 4x reuse of the f32 A rows
//        in the P@V GEMM, which otherwise streams 8 GB from HBM).
//    A: f16 row-major (f32 converted in-register when A_F32).
//    Bt: f16 row-major [N,K] (B transposed -> per-lane contiguous K).
//    Lane layouts follow CDNA5 ISA 7.12.2.
//    Requires M % 32 == 0, N % 64 == 0, K % 32 == 0 (true for all calls).
// ---------------------------------------------------------------------------
#define BM 2
#define BN 4

template <bool A_F32, bool OUT_F16, bool OUT_TRANS>
__global__ void wmma_gemm(const void* __restrict__ A_,
                          const _Float16* __restrict__ Bt,
                          const float* __restrict__ bias,
                          float scale,
                          void* __restrict__ C_,
                          int M, int N, int K, int ldc)
{
    const int lane = threadIdx.x & 31;
    const int wave = blockIdx.x * (blockDim.x >> 5) + (threadIdx.x >> 5);
    const int tilesN = N >> 6;                       // 64-wide blocks
    const int tiles  = (M >> 5) * tilesN;            // 32-tall blocks
    if (wave >= tiles) return;                       // wave-uniform: EXEC all-1s
    const int tm = wave / tilesN;
    const int tn = wave - tm * tilesN;
    const int row0 = tm << 5;
    const int col0 = tn << 6;

    const int r  = lane & 15;                        // row (A) / column (B) within tile
    const int hi = lane >> 4;                        // lane half selects K sub-ranges
    const int kbA = hi ? 8 : 0;                      // A: K in [kbA,kbA+8) and [kbA+16,kbA+24)
    const int kbB = hi ? 8 : 0;                      // B: K in [kbB,kbB+8) and [kbB+16,kbB+24)

    v8f acc[BM][BN];
#pragma unroll
    for (int i = 0; i < BM; ++i)
#pragma unroll
        for (int j = 0; j < BN; ++j) acc[i][j] = (v8f){};

    for (int k0 = 0; k0 < K; k0 += 32) {
        v16h a[BM], b[BN];
#pragma unroll
        for (int i = 0; i < BM; ++i) {
            if (A_F32) {
                const float* pa = (const float*)A_ + (size_t)(row0 + 16 * i + r) * K + k0 + kbA;
#pragma unroll
                for (int u = 0; u < 8; ++u) { a[i][u] = (_Float16)pa[u]; a[i][u + 8] = (_Float16)pa[16 + u]; }
            } else {
                const _Float16* pa = (const _Float16*)A_ + (size_t)(row0 + 16 * i + r) * K + k0 + kbA;
                const v8h* p0 = reinterpret_cast<const v8h*>(pa);
                const v8h* p1 = reinterpret_cast<const v8h*>(pa + 16);
#pragma unroll
                for (int u = 0; u < 8; ++u) { a[i][u] = (*p0)[u]; a[i][u + 8] = (*p1)[u]; }
            }
        }
#pragma unroll
        for (int j = 0; j < BN; ++j) {
            const _Float16* pb = Bt + (size_t)(col0 + 16 * j + r) * K + k0 + kbB;
            const v8h* q0 = reinterpret_cast<const v8h*>(pb);
            const v8h* q1 = reinterpret_cast<const v8h*>(pb + 16);
#pragma unroll
            for (int u = 0; u < 8; ++u) { b[j][u] = (*q0)[u]; b[j][u + 8] = (*q1)[u]; }
        }
#pragma unroll
        for (int i = 0; i < BM; ++i)
#pragma unroll
            for (int j = 0; j < BN; ++j)
                acc[i][j] = __builtin_amdgcn_wmma_f32_16x16x32_f16(
                    /*neg_a=*/false, a[i], /*neg_b=*/false, b[j],
                    /*c_mod=*/(short)0, acc[i][j],
                    /*reuse_a=*/false, /*reuse_b=*/false);
    }

    // C/D layout: VGPR u -> M = u + 8*hi ; N = lane & 15
    const int nlane = lane & 15;
    const int mbase = hi ? 8 : 0;
#pragma unroll
    for (int i = 0; i < BM; ++i) {
#pragma unroll
        for (int j = 0; j < BN; ++j) {
            const int cn = col0 + 16 * j + nlane;
            const float bv = bias ? bias[cn] : 0.0f;
#pragma unroll
            for (int u = 0; u < 8; ++u) {
                float v = acc[i][j][u] * scale + bv;
                int m = row0 + 16 * i + mbase + u;
                if (OUT_TRANS) {
                    if (OUT_F16) ((_Float16*)C_)[(size_t)cn * ldc + m] = (_Float16)v;
                    else         ((float*)   C_)[(size_t)cn * ldc + m] = v;
                } else {
                    if (OUT_F16) ((_Float16*)C_)[(size_t)m * ldc + cn] = (_Float16)v;
                    else         ((float*)   C_)[(size_t)m * ldc + cn] = v;
                }
            }
        }
    }
}

// ---------------------------------------------------------------------------
// 4) LSTM recurrent scan. grid = 2 (dir 0 fwd, dir 1 rev), block = 1024.
//    pre[t,g] already holds x_t @ W_ih.T + b_ih.  h lives in LDS; W_hh (1 MB)
//    stays L2-resident across the 8192 steps.
// ---------------------------------------------------------------------------
__global__ void __launch_bounds__(1024)
lstm_scan(const float* __restrict__ pre_f, const float* __restrict__ pre_r,
          const float* __restrict__ whh_f, const float* __restrict__ whh_r,
          const float* __restrict__ bhh_f, const float* __restrict__ bhh_r,
          float* __restrict__ lstm_out)            // [N, 512]
{
    __shared__ float h_sh[H_LSTM];
    __shared__ float gates[G4H];
    const int dir = blockIdx.x;
    const float* pre = dir ? pre_r : pre_f;
    const float* whh = dir ? whh_r : whh_f;
    const float* bhh = dir ? bhh_r : bhh_f;
    const int tid = threadIdx.x;

    if (tid < H_LSTM) h_sh[tid] = 0.0f;
    float c = 0.0f;
    const float bh = bhh[tid];
    const float* wr = whh + (size_t)tid * H_LSTM;
    __syncthreads();

    for (int s = 0; s < NN; ++s) {
        const int t = dir ? (NN - 1 - s) : s;
        float g = pre[(size_t)t * G4H + tid] + bh;
#pragma unroll 8
        for (int k = 0; k < H_LSTM; ++k) g += wr[k] * h_sh[k];
        gates[tid] = g;
        __syncthreads();
        if (tid < H_LSTM) {
            float gi = 1.0f / (1.0f + __expf(-gates[tid]));
            float gf = 1.0f / (1.0f + __expf(-gates[H_LSTM + tid]));
            float gg = tanhf(gates[2 * H_LSTM + tid]);
            float go = 1.0f / (1.0f + __expf(-gates[3 * H_LSTM + tid]));
            c = gf * c + gi * gg;
            float h = go * tanhf(c);
            h_sh[tid] = h;
            lstm_out[(size_t)t * D_ATT + dir * H_LSTM + tid] = h;
        }
        __syncthreads();
    }
}

// ---------------------------------------------------------------------------
// 5) Row softmax in place (one 256-thread block per row of 8192)
// ---------------------------------------------------------------------------
__global__ void softmax_rows(float* __restrict__ a, int ncols)
{
    __shared__ float sm[256];
    float* p = a + (size_t)blockIdx.x * ncols;
    const int tid = threadIdx.x;

    float m = -1e30f;
    for (int c = tid; c < ncols; c += 256) m = fmaxf(m, p[c]);
    sm[tid] = m; __syncthreads();
    for (int s = 128; s > 0; s >>= 1) {
        if (tid < s) sm[tid] = fmaxf(sm[tid], sm[tid + s]);
        __syncthreads();
    }
    m = sm[0]; __syncthreads();

    float sum = 0.0f;
    for (int c = tid; c < ncols; c += 256) {
        float e = __expf(p[c] - m);
        p[c] = e;
        sum += e;
    }
    sm[tid] = sum; __syncthreads();
    for (int s = 128; s > 0; s >>= 1) {
        if (tid < s) sm[tid] += sm[tid + s];
        __syncthreads();
    }
    float inv = 1.0f / sm[0];
    __syncthreads();
    for (int c = tid; c < ncols; c += 256) p[c] *= inv;
}

// ---------------------------------------------------------------------------
// 6) FC head: out[n] = dot(proj[n,:512], fc_w) + fc_b
// ---------------------------------------------------------------------------
__global__ void fc_head(const float* __restrict__ proj,
                        const float* __restrict__ fcw,
                        const float* __restrict__ fcb,
                        float* __restrict__ out)
{
    int n = blockIdx.x * blockDim.x + threadIdx.x;
    if (n >= NN) return;
    const float* p = proj + (size_t)n * D_ATT;
    float acc = fcb[0];
#pragma unroll 8
    for (int k = 0; k < D_ATT; ++k) acc += fcw[k] * p[k];
    out[n] = acc;
}

// ---------------------------------------------------------------------------
// Host-side orchestration
// ---------------------------------------------------------------------------
static inline int gemm_blocks(int M, int N) {
    int tiles = (M >> 5) * (N >> 6);               // 32x64 per wave
    return (tiles + 7) / 8;                        // 8 waves (256 threads) per block
}

extern "C" void kernel_launch(void* const* d_in, const int* in_sizes, int n_in,
                              void* d_out, int out_size, void* d_ws, size_t ws_size,
                              hipStream_t stream)
{
    (void)in_sizes; (void)n_in; (void)out_size; (void)ws_size;
    const float* x        = (const float*)d_in[0];
    const int*   ei       = (const int*)  d_in[1];
    const float* ea       = (const float*)d_in[2];
    const float* msg_w    = (const float*)d_in[3];
    const float* msg_b    = (const float*)d_in[4];
    const float* upd_w    = (const float*)d_in[5];
    const float* upd_b    = (const float*)d_in[6];
    const float* w_ih_f   = (const float*)d_in[7];
    const float* w_hh_f   = (const float*)d_in[8];
    const float* b_ih_f   = (const float*)d_in[9];
    const float* b_hh_f   = (const float*)d_in[10];
    const float* w_ih_r   = (const float*)d_in[11];
    const float* w_hh_r   = (const float*)d_in[12];
    const float* b_ih_r   = (const float*)d_in[13];
    const float* b_hh_r   = (const float*)d_in[14];
    const float* in_pw    = (const float*)d_in[15];
    const float* in_pb    = (const float*)d_in[16];
    const float* out_pw   = (const float*)d_in[17];
    const float* out_pb   = (const float*)d_in[18];
    const float* fc_w     = (const float*)d_in[19];
    const float* fc_b     = (const float*)d_in[20];

    float* fc_out = (float*)d_out;                 // [N]
    float* attnW  = (float*)d_out + NN;            // [N,N] scores -> softmax in place

    // ---- workspace bump allocator (256B aligned) ----
    char* wp = (char*)d_ws;
    auto alloc = [&](size_t bytes) -> void* {
        void* p = (void*)wp;
        wp += (bytes + 255) & ~(size_t)255;
        return p;
    };
    float*    aggr    = (float*)   alloc((size_t)NN * D_MSG * 4);
    _Float16* h16     = (_Float16*)alloc((size_t)NN * D_NODE * 2);
    _Float16* wihf16  = (_Float16*)alloc((size_t)G4H * D_NODE * 2);
    _Float16* wihr16  = (_Float16*)alloc((size_t)G4H * D_NODE * 2);
    _Float16* qkv16   = (_Float16*)alloc((size_t)3 * D_ATT * D_ATT * 2);
    _Float16* wo16    = (_Float16*)alloc((size_t)D_ATT * D_ATT * 2);
    float*    pre_f   = (float*)   alloc((size_t)NN * G4H * 4);
    float*    pre_r   = (float*)   alloc((size_t)NN * G4H * 4);
    float*    lstm    = (float*)   alloc((size_t)NN * D_ATT * 4);
    _Float16* lstm16  = (_Float16*)alloc((size_t)NN * D_ATT * 2);
    _Float16* q16     = (_Float16*)alloc((size_t)NN * D_ATT * 2);
    _Float16* k16     = (_Float16*)alloc((size_t)NN * D_ATT * 2);
    _Float16* vt16    = (_Float16*)alloc((size_t)D_ATT * NN * 2);   // V transposed
    _Float16* a16     = (_Float16*)alloc((size_t)NN * D_ATT * 2);
    float*    proj    = (float*)   alloc((size_t)NN * D_ATT * 4);

    // ---- 1) MPNN ----
    hipMemsetAsync(aggr, 0, (size_t)NN * D_MSG * 4, stream);
    {
        long th = (long)EE * D_MSG;
        mpnn_msg_scatter<<<(int)((th + 255) / 256), 256, 0, stream>>>(x, ei, ea, msg_w, msg_b, aggr);
    }
    {
        long th = (long)NN * D_NODE;
        mpnn_update<<<(int)((th + 255) / 256), 256, 0, stream>>>(x, aggr, upd_w, upd_b, h16);
    }

    // ---- weight conversions to f16 ----
    cvt_f32_f16<<<(G4H * D_NODE + 255) / 256, 256, 0, stream>>>(w_ih_f, wihf16, (long)G4H * D_NODE);
    cvt_f32_f16<<<(G4H * D_NODE + 255) / 256, 256, 0, stream>>>(w_ih_r, wihr16, (long)G4H * D_NODE);
    cvt_f32_f16<<<(3 * D_ATT * D_ATT + 255) / 256, 256, 0, stream>>>(in_pw, qkv16, (long)3 * D_ATT * D_ATT);
    cvt_f32_f16<<<(D_ATT * D_ATT + 255) / 256, 256, 0, stream>>>(out_pw, wo16, (long)D_ATT * D_ATT);

    // ---- 2) LSTM input projections (WMMA GEMMs), then the serial scan ----
    wmma_gemm<false, false, false><<<gemm_blocks(NN, G4H), 256, 0, stream>>>(
        h16, wihf16, b_ih_f, 1.0f, pre_f, NN, G4H, D_NODE, G4H);
    wmma_gemm<false, false, false><<<gemm_blocks(NN, G4H), 256, 0, stream>>>(
        h16, wihr16, b_ih_r, 1.0f, pre_r, NN, G4H, D_NODE, G4H);
    lstm_scan<<<2, 1024, 0, stream>>>(pre_f, pre_r, w_hh_f, w_hh_r, b_hh_f, b_hh_r, lstm);
    cvt_f32_f16<<<(int)(((long)NN * D_ATT + 255) / 256), 256, 0, stream>>>(lstm, lstm16, (long)NN * D_ATT);

    // ---- 3) Attention ----
    const _Float16* wq16 = qkv16;
    const _Float16* wk16 = qkv16 + (size_t)D_ATT * D_ATT;
    const _Float16* wv16 = qkv16 + (size_t)2 * D_ATT * D_ATT;
    const float* bq = in_pb;
    const float* bk = in_pb + D_ATT;
    const float* bv = in_pb + 2 * D_ATT;

    wmma_gemm<false, true, false><<<gemm_blocks(NN, D_ATT), 256, 0, stream>>>(
        lstm16, wq16, bq, 1.0f, q16, NN, D_ATT, D_ATT, D_ATT);
    wmma_gemm<false, true, false><<<gemm_blocks(NN, D_ATT), 256, 0, stream>>>(
        lstm16, wk16, bk, 1.0f, k16, NN, D_ATT, D_ATT, D_ATT);
    wmma_gemm<false, true, true><<<gemm_blocks(NN, D_ATT), 256, 0, stream>>>(
        lstm16, wv16, bv, 1.0f, vt16, NN, D_ATT, D_ATT, NN);    // transposed out, ldc = N rows

    // scores = Q @ K^T / sqrt(512)   (K matrix row-major already == Bt layout)
    const float inv_sqrt_d = 1.0f / sqrtf((float)D_ATT);
    wmma_gemm<false, false, false><<<gemm_blocks(NN, NN), 256, 0, stream>>>(
        q16, k16, nullptr, inv_sqrt_d, attnW, NN, NN, D_ATT, NN);

    softmax_rows<<<NN, 256, 0, stream>>>(attnW, NN);

    // attn_out = P @ V   (A = f32 probabilities converted in-register; Bt = V^T)
    wmma_gemm<true, true, false><<<gemm_blocks(NN, D_ATT), 256, 0, stream>>>(
        attnW, vt16, nullptr, 1.0f, a16, NN, D_ATT, NN, D_ATT);

    // out projection
    wmma_gemm<false, false, false><<<gemm_blocks(NN, D_ATT), 256, 0, stream>>>(
        a16, wo16, out_pb, 1.0f, proj, NN, D_ATT, D_ATT, D_ATT);

    // ---- 4) FC head ----
    fc_head<<<(NN + 255) / 256, 256, 0, stream>>>(proj, fc_w, fc_b, fc_out);
}
